// EquivariantUpdate_72060961292772
// MI455X (gfx1250) — matrix operand order repaired and checked by reading, not compiled, not run on hardware
//
#include <hip/hip_runtime.h>
#include <hip/hip_bf16.h>

typedef __attribute__((ext_vector_type(16))) _Float16 v16h;
typedef __attribute__((ext_vector_type(8)))  _Float16 v8h;
typedef __attribute__((ext_vector_type(8)))  float    v8f;

#define HIDDEN   128
#define NEDGES   800000
#define NNODES   50000
#define INV_NORM 0.01f

// ---------------- workspace layout (halves) ----------------
// h16    : NNODES*HIDDEN                     (6,400,000 halves)
// w1frag : 8 ktiles * 8 ntiles * 512 halves  (32,768 halves)
// w2frag : 4 ktiles * 8 ntiles * 512 halves  (16,384 halves)
#define H16_OFF   0
#define W1F_OFF   (NNODES * HIDDEN)
#define W2F_OFF   (W1F_OFF + 8 * 8 * 512)

// ---------------- prep kernels ----------------
__global__ void cvt_h_kernel(const float* __restrict__ h, _Float16* __restrict__ h16, int n) {
    int i = blockIdx.x * blockDim.x + threadIdx.x;
    if (i < n) h16[i] = (_Float16)h[i];
}

// Pack W[K x 128] into per-lane WMMA B fragments:
// fragment f = kt*8 + nt; element index = (f*32 + lane)*16 + half
// lane<16 : N = nt*16 + lane,    K = kt*32 + (half<8 ? half : 16 + half-8)
// lane>=16: N = nt*16 + lane-16, K = kt*32 + 8 + (half<8 ? half : 16 + half-8)
__global__ void pack_w_kernel(const float* __restrict__ W, _Float16* __restrict__ frag, int ktiles) {
    int i = blockIdx.x * blockDim.x + threadIdx.x;
    int total = ktiles * 8 * 512;
    if (i >= total) return;
    int f      = i >> 9;
    int within = i & 511;
    int lane   = within >> 4;
    int half   = within & 15;
    int kt = f >> 3;
    int nt = f & 7;
    int K = kt * 32 + ((half & 8) << 1) + ((lane >= 16) ? 8 : 0) + (half & 7);
    int N = nt * 16 + (lane & 15);
    frag[i] = (_Float16)W[K * HIDDEN + N];
}

__global__ void init_out_kernel(const float* __restrict__ coord, float* __restrict__ out, int n) {
    int i = blockIdx.x * blockDim.x + threadIdx.x;
    if (i < n) out[i] = coord[i];
}

__global__ void fin_out_kernel(float* __restrict__ out, const float* __restrict__ node_mask, int n) {
    int i = blockIdx.x * blockDim.x + threadIdx.x;
    if (i < n) out[i] = out[i] * node_mask[i / 3];
}

// ---------------- main fused edge-MLP kernel ----------------
// 128 threads = 4 waves; each wave owns TWO 16-edge tiles (32 edges),
// so every B fragment load feeds two WMMAs.
__global__ __launch_bounds__(128) void egnn_edge_kernel(
    const int*   __restrict__ ei,          // [2, E]
    const float* __restrict__ edge_attr,   // [E, 1]
    const float* __restrict__ edge_mask,   // [E, 1]
    const float* __restrict__ coord_diff,  // [E, 3]
    const float* __restrict__ W1,          // [257, 128] raw (for last row)
    const float* __restrict__ b1,          // [128]
    const float* __restrict__ b2,          // [128]
    const float* __restrict__ W3,          // [128, 1]
    const _Float16* __restrict__ h16,      // [NNODES, 128]
    const _Float16* __restrict__ w1f,      // packed B frags, 8 ktiles
    const _Float16* __restrict__ w2f,      // packed B frags, 4 ktiles
    float* __restrict__ out)               // [NNODES, 3], pre-init to coord
{
    __shared__ _Float16 lds_x[4][32 * 136];   // per-wave 32x128 f16 tile, padded stride
    __shared__ float    lds_ea[4][32];

    const int lane = threadIdx.x & 31;
    const int wv   = threadIdx.x >> 5;
    const int tile = blockIdx.x * 4 + wv;
    const int eb   = tile * 32;

    const int mA   = lane & 15;                 // this lane's edge row within an M-tile
    const int koff = (lane >= 16) ? 8 : 0;      // A layout hi-lane K shift / C-row shift
    const int nloc = lane & 15;

    int r[2], c[2];
#pragma unroll
    for (int t = 0; t < 2; ++t) {
        const int eA = min(eb + t * 16 + mA, NEDGES - 1);
        r[t] = ei[eA];
        c[t] = ei[NEDGES + eA];
        if (lane < 16) lds_ea[wv][t * 16 + lane] = edge_attr[eA];
    }
    __syncthreads();

    // ---------------- layer 1: [32 x 256] x [256 x 128] ----------------
    v8f acc[2][8] = {};
    const v16h* w1v = (const v16h*)w1f;
#pragma unroll
    for (int kt = 0; kt < 8; ++kt) {
        const int k0 = kt * 32 + koff;          // halves 0..7  -> K = k0..k0+7
        const int k1 = kt * 32 + 16 + koff;     // halves 8..15 -> K = k1..k1+7
        v16h a[2];
#pragma unroll
        for (int t = 0; t < 2; ++t) {
            const _Float16* hr = h16 + (size_t)r[t] * HIDDEN;
            const _Float16* hc = h16 + (size_t)c[t] * HIDDEN;
            const _Float16* p0 = (k0 < 128) ? (hr + k0) : (hc + (k0 - 128));
            const _Float16* p1 = (k1 < 128) ? (hr + k1) : (hc + (k1 - 128));
            const v8h lo = *(const v8h*)p0;
            const v8h hi = *(const v8h*)p1;
#pragma unroll
            for (int i = 0; i < 8; ++i) { a[t][i] = lo[i]; a[t][8 + i] = hi[i]; }
        }
        const v16h* bf = w1v + (size_t)(kt * 8) * 32 + lane;
#pragma unroll
        for (int nt = 0; nt < 8; ++nt) {
            v16h b = bf[nt * 32];
            acc[0][nt] = __builtin_amdgcn_wmma_f32_16x16x32_f16(
                false, a[0], false, b, (short)0, acc[0][nt], false, false);
            acc[1][nt] = __builtin_amdgcn_wmma_f32_16x16x32_f16(
                false, a[1], false, b, (short)0, acc[1][nt], false, false);
        }
    }

    // epilogue 1: + b1 + edge_attr * W1[256,:], SiLU, stash f16 tile to LDS
    float eav[2][8];
#pragma unroll
    for (int t = 0; t < 2; ++t)
#pragma unroll
        for (int v = 0; v < 8; ++v) eav[t][v] = lds_ea[wv][t * 16 + v + koff];
    const float* w1last = W1 + 256 * HIDDEN;
    _Float16* xrow = &lds_x[wv][0];
#pragma unroll
    for (int nt = 0; nt < 8; ++nt) {
        const int   N  = nt * 16 + nloc;
        const float bb = b1[N];
        const float wl = w1last[N];
#pragma unroll
        for (int t = 0; t < 2; ++t) {
#pragma unroll
            for (int v = 0; v < 8; ++v) {
                float val = acc[t][nt][v] + bb + eav[t][v] * wl;
                val = val / (1.0f + __expf(-val));          // SiLU
                xrow[(t * 16 + v + koff) * 136 + N] = (_Float16)val;
            }
        }
    }
    __syncthreads();

    // ---------------- layer 2: [32 x 128] x [128 x 128] ----------------
#pragma unroll
    for (int t = 0; t < 2; ++t)
#pragma unroll
        for (int nt = 0; nt < 8; ++nt) acc[t][nt] = (v8f){};
    const v16h* w2v = (const v16h*)w2f;
#pragma unroll
    for (int kt = 0; kt < 4; ++kt) {
        const int k0 = kt * 32 + koff;
        const int k1 = kt * 32 + 16 + koff;
        v16h a[2];
#pragma unroll
        for (int t = 0; t < 2; ++t) {
            const _Float16* arow = &lds_x[wv][(t * 16 + mA) * 136];
            const v8h lo = *(const v8h*)(arow + k0);
            const v8h hi = *(const v8h*)(arow + k1);
#pragma unroll
            for (int i = 0; i < 8; ++i) { a[t][i] = lo[i]; a[t][8 + i] = hi[i]; }
        }
        const v16h* bf = w2v + (size_t)(kt * 8) * 32 + lane;
#pragma unroll
        for (int nt = 0; nt < 8; ++nt) {
            v16h b = bf[nt * 32];
            acc[0][nt] = __builtin_amdgcn_wmma_f32_16x16x32_f16(
                false, a[0], false, b, (short)0, acc[0][nt], false, false);
            acc[1][nt] = __builtin_amdgcn_wmma_f32_16x16x32_f16(
                false, a[1], false, b, (short)0, acc[1][nt], false, false);
        }
    }

    // ---------------- layer 3: x2 @ W3, SiLU fused, per-edge scalar ----------------
    float part[2][8];
#pragma unroll
    for (int t = 0; t < 2; ++t)
#pragma unroll
        for (int v = 0; v < 8; ++v) part[t][v] = 0.0f;
#pragma unroll
    for (int nt = 0; nt < 8; ++nt) {
        const int   N   = nt * 16 + nloc;
        const float bb  = b2[N];
        const float w3n = W3[N];
#pragma unroll
        for (int t = 0; t < 2; ++t) {
#pragma unroll
            for (int v = 0; v < 8; ++v) {
                float val = acc[t][nt][v] + bb;
                val = val / (1.0f + __expf(-val));          // SiLU
                part[t][v] += val * w3n;
            }
        }
    }
    // reduce across the 16-lane half-wave that shares each output row
#pragma unroll
    for (int off = 1; off <= 8; off <<= 1) {
#pragma unroll
        for (int t = 0; t < 2; ++t)
#pragma unroll
            for (int v = 0; v < 8; ++v)
                part[t][v] += __shfl_xor(part[t][v], off, 32);
    }

    // scatter: lane (v) of each half handles edge m = v + koff of each tile
    const int v = nloc;
    if (v < 8) {
        const int m = v + koff;
#pragma unroll
        for (int t = 0; t < 2; ++t) {
            const int e = eb + t * 16 + m;
            if (e < NEDGES) {
                const float s  = part[t][v] * edge_mask[e] * INV_NORM;
                const int   rr = ei[e];
                atomicAdd(&out[rr * 3 + 0], coord_diff[e * 3 + 0] * s);
                atomicAdd(&out[rr * 3 + 1], coord_diff[e * 3 + 1] * s);
                atomicAdd(&out[rr * 3 + 2], coord_diff[e * 3 + 2] * s);
            }
        }
    }
}

// ---------------- launcher ----------------
extern "C" void kernel_launch(void* const* d_in, const int* in_sizes, int n_in,
                              void* d_out, int out_size, void* d_ws, size_t ws_size,
                              hipStream_t stream) {
    // inputs: 0 h, 1 coord, 2 edge_index, 3 coord_diff, 4 edge_attr, 5 edge_mask,
    //         6 node_mask, 7 W1, 8 b1, 9 W2, 10 b2, 11 W3
    const float* h          = (const float*)d_in[0];
    const float* coord      = (const float*)d_in[1];
    const int*   ei         = (const int*)  d_in[2];
    const float* coord_diff = (const float*)d_in[3];
    const float* edge_attr  = (const float*)d_in[4];
    const float* edge_mask  = (const float*)d_in[5];
    const float* node_mask  = (const float*)d_in[6];
    const float* W1         = (const float*)d_in[7];
    const float* b1         = (const float*)d_in[8];
    const float* W2         = (const float*)d_in[9];
    const float* b2         = (const float*)d_in[10];
    const float* W3         = (const float*)d_in[11];
    float* out = (float*)d_out;

    _Float16* ws16 = (_Float16*)d_ws;
    _Float16* h16  = ws16 + H16_OFF;
    _Float16* w1f  = ws16 + W1F_OFF;
    _Float16* w2f  = ws16 + W2F_OFF;

    const int nh = NNODES * HIDDEN;
    cvt_h_kernel<<<(nh + 255) / 256, 256, 0, stream>>>(h, h16, nh);
    pack_w_kernel<<<(8 * 8 * 512 + 255) / 256, 256, 0, stream>>>(W1, w1f, 8);
    pack_w_kernel<<<(4 * 8 * 512 + 255) / 256, 256, 0, stream>>>(W2, w2f, 4);

    const int nc = NNODES * 3;
    init_out_kernel<<<(nc + 255) / 256, 256, 0, stream>>>(coord, out, nc);

    const int tiles  = (NEDGES + 31) / 32;      // 25000 double-tiles
    const int blocks = (tiles + 3) / 4;         // 6250
    egnn_edge_kernel<<<blocks, 128, 0, stream>>>(
        ei, edge_attr, edge_mask, coord_diff, W1, b1, b2, W3,
        h16, w1f, w2f, out);

    fin_out_kernel<<<(nc + 255) / 256, 256, 0, stream>>>(out, node_mask, nc);
}
